// multi_dRNN_with_dilations_25417616458215
// MI455X (gfx1250) — compile-verified, matrix operand order checked by reading, not run on hardware
//
#include <hip/hip_runtime.h>
#include <hip/hip_bf16.h>
#include <stdint.h>

// ---------------------------------------------------------------------------
// CDNA5 (gfx1250) dilated-LSTM stack, round 2.
//  * input projections: 32x64-per-wave bf16 WMMA GEMM, register double-buffered
//  * recurrence: one persistent kernel per layer (grid-wide atomic barrier per
//    timestep), cell state held in registers, h_{t-1} staged to LDS via
//    global_load_async_to_lds_b128 (ASYNCcnt), A fragments fed from LDS.
// ---------------------------------------------------------------------------

typedef __attribute__((ext_vector_type(16))) __bf16       v16bf;
typedef __attribute__((ext_vector_type(8)))  float        v8f;
typedef __attribute__((ext_vector_type(4)))  unsigned int v4u;

#define DEV_INLINE __device__ __forceinline__

DEV_INLINE unsigned short f2bf(float f) {
  unsigned int u = __float_as_uint(f);
  unsigned int r = (u + 0x7FFFu + ((u >> 16) & 1u)) >> 16;  // RNE
  return (unsigned short)r;
}

union FragBF { v4u q[2]; v16bf v; };

// 32-byte fragment fetch: q0 = K[k..k+7], q1 = K[k+16..k+23] (per-lane K-half
// selection is folded into the base pointer by the caller). ISA 7.12.2 layout.
DEV_INLINE void load2(FragBF& f, const unsigned short* p) {
  f.q[0] = *(const v4u*)(p);
  f.q[1] = *(const v4u*)(p + 16);
}

DEV_INLINE float sigmoidf_(float x) { return 1.0f / (1.0f + __expf(-x)); }

#define WMMA_BF16(A, B, C) \
  __builtin_amdgcn_wmma_f32_16x16x32_bf16(false, (A), false, (B), (short)0, (C), false, false)

// gfx1250 async direct-to-LDS copy (16B per lane), tracked by ASYNCcnt.
DEV_INLINE void async_g2l_b128(unsigned lds_off, const void* gaddr) {
  asm volatile("global_load_async_to_lds_b128 %0, %1, off"
               :: "v"(lds_off), "v"((unsigned long long)(uintptr_t)gaddr)
               : "memory");
}
DEV_INLINE void wait_async0() { asm volatile("s_wait_asynccnt 0" ::: "memory"); }

// ---------------------------------------------------------------------------
// Elementwise helpers
// ---------------------------------------------------------------------------
__global__ void k_f32_to_bf16(const float* __restrict__ src,
                              unsigned short* __restrict__ dst, int n) {
  int i = blockIdx.x * blockDim.x + threadIdx.x;
  if (i < n) dst[i] = f2bf(src[i]);
}

__global__ void k_bias_sum(const float* __restrict__ a,
                           const float* __restrict__ b,
                           float* __restrict__ o, int n) {
  int i = blockIdx.x * blockDim.x + threadIdx.x;
  if (i < n) o[i] = a[i] + b[i];
}

__global__ void k_zero_u32(unsigned* __restrict__ p, int n) {
  int i = blockIdx.x * blockDim.x + threadIdx.x;
  if (i < n) p[i] = 0u;
}

// ---------------------------------------------------------------------------
// Input projection GEMM: G[R,2048] = X[R,512](bf16) @ W[2048,512]^T + bias.
// Block = 256 (8 waves). Each wave: 32 rows x 64 cols = 8 accumulator tiles,
// A pair reused across 4 B tiles, all fragments register double-buffered so
// iteration k+1's loads overlap iteration k's 8 WMMAs. grid = (R/32, 4).
// ---------------------------------------------------------------------------
__global__ __launch_bounds__(256)
void k_gemm_xw(const unsigned short* __restrict__ X,
               const unsigned short* __restrict__ W,
               const float* __restrict__ bias,
               float* __restrict__ G, int R) {
  const int lane  = threadIdx.x & 31;
  const int wave  = threadIdx.x >> 5;
  const int lr    = lane & 15;
  const int khalf = (lane >> 4) << 3;   // 0 / 8 : per-lane K-half
  const int hi8   = khalf;              // C/D row offset for lanes 16-31
  const int m0    = blockIdx.x << 5;                   // 32 rows
  const int cb    = (blockIdx.y << 9) + (wave << 6);   // 64 cols

  v8f acc[8];
#pragma unroll
  for (int i = 0; i < 8; ++i) acc[i] = (v8f){};

  const unsigned short* arow0 = X + (size_t)(m0 + lr) * 512 + khalf;
  const unsigned short* arow1 = arow0 + 16 * 512;
  const unsigned short* br    = W + (size_t)(cb + lr) * 512 + khalf;

  FragBF a0c, a1c, b0c, b1c, b2c, b3c;
  FragBF a0n, a1n, b0n, b1n, b2n, b3n;
  load2(a0c, arow0);            load2(a1c, arow1);
  load2(b0c, br);               load2(b1c, br + 16 * 512);
  load2(b2c, br + 32 * 512);    load2(b3c, br + 48 * 512);

#pragma unroll 4
  for (int k = 0; k < 512; k += 32) {
    const int kn = k + 32;
    if (kn < 512) {  // prefetch next K-chunk into the shadow register set
      load2(a0n, arow0 + kn);         load2(a1n, arow1 + kn);
      load2(b0n, br + kn);            load2(b1n, br + 16 * 512 + kn);
      load2(b2n, br + 32 * 512 + kn); load2(b3n, br + 48 * 512 + kn);
    }
    acc[0] = WMMA_BF16(a0c.v, b0c.v, acc[0]);
    acc[1] = WMMA_BF16(a0c.v, b1c.v, acc[1]);
    acc[2] = WMMA_BF16(a0c.v, b2c.v, acc[2]);
    acc[3] = WMMA_BF16(a0c.v, b3c.v, acc[3]);
    acc[4] = WMMA_BF16(a1c.v, b0c.v, acc[4]);
    acc[5] = WMMA_BF16(a1c.v, b1c.v, acc[5]);
    acc[6] = WMMA_BF16(a1c.v, b2c.v, acc[6]);
    acc[7] = WMMA_BF16(a1c.v, b3c.v, acc[7]);
    a0c = a0n; a1c = a1n; b0c = b0n; b1c = b1n; b2c = b2n; b3c = b3n;
  }

#pragma unroll
  for (int rh = 0; rh < 2; ++rh) {
    float* gout = G + (size_t)(m0 + rh * 16 + hi8) * 2048;
#pragma unroll
    for (int tile = 0; tile < 4; ++tile) {
      const int col  = cb + tile * 16 + lr;
      const float bv = bias[col];
#pragma unroll
      for (int i = 0; i < 8; ++i)
        gout[(size_t)i * 2048 + col] = acc[rh * 4 + tile][i] + bv;
    }
  }
}

// ---------------------------------------------------------------------------
// Persistent per-layer LSTM scan. grid = (N/16, 32), block = 128 (4 waves).
// Block (bx,by) owns rows bx*16.. and hidden cols by*16.. for ALL timesteps:
//   - cell state stays in registers (2 floats / thread)
//   - wave g computes gate g's 16x16 tile: acc = Xg + h_{t-1} @ Whh^T tile
//   - h_{t-1} slab (16x512 bf16) staged once per block per step into LDS via
//     async direct-to-LDS loads; A fragments then come from LDS (ds_load_b128)
//   - gate tiles meet in LDS for the fused cell update
//   - timesteps separated by a device-scope atomic barrier (monotonic counter)
// ---------------------------------------------------------------------------
__global__ __launch_bounds__(128)
void k_lstm_layer(const float* __restrict__ G,            // [16384,2048]
                  const unsigned short* __restrict__ Whh, // [2048,512] bf16
                  unsigned short* __restrict__ H,         // [16384,512] bf16
                  float* __restrict__ Fout,               // fp32 out or nullptr
                  unsigned* __restrict__ bar,             // zero-initialized
                  int N, int Td) {
  __shared__ unsigned short hA[16 * 512];  // 16 KB h_{t-1} slab
  __shared__ float gb[4][16][16];          // 4 KB gate exchange
  const int tid   = threadIdx.x;
  const int lane  = tid & 31;
  const int gate  = tid >> 5;
  const int lr    = lane & 15;
  const int khalf = (lane >> 4) << 3;
  const int hi8   = khalf;
  const int m0    = blockIdx.x << 4;
  const int n0    = blockIdx.y << 4;
  const unsigned nblocks = gridDim.x * gridDim.y;

  // pointwise assignment: 2 of the 256 tile elements per thread
  const int pm0 = tid >> 4,         pc0 = tid & 15;
  const int pm1 = (tid + 128) >> 4, pc1 = (tid + 128) & 15;
  float c0 = 0.0f, c1 = 0.0f;  // cell state in registers (c_0 = 0)

  const unsigned short* brow = Whh + (size_t)(gate * 512 + n0 + lr) * 512 + khalf;
  const unsigned short* aldsrow = &hA[lr * 512 + khalf];
  const unsigned hA_base = (unsigned)(uintptr_t)(&hA[0]);  // LDS byte offset

  for (int t = 0; t < Td; ++t) {
    const size_t rowbase = (size_t)t * N;

    // Seed accumulator with the precomputed input projection (+fused bias).
    v8f acc;
    {
      const float* gp =
          G + (rowbase + m0 + hi8) * 2048 + (size_t)gate * 512 + n0 + lr;
#pragma unroll
      for (int i = 0; i < 8; ++i) acc[i] = gp[(size_t)i * 2048];
    }

    if (t > 0) {
      // Stage h_{t-1} rows [m0, m0+16) x 512 into LDS: 1024 x 16B chunks,
      // 8 async lane-copies per thread, drained on ASYNCcnt.
      const char* hsrc = (const char*)(H + (rowbase - (size_t)N + m0) * 512);
#pragma unroll
      for (int i = 0; i < 8; ++i) {
        const int chunk = tid + i * 128;
        async_g2l_b128(hA_base + chunk * 16, hsrc + chunk * 16);
      }
      wait_async0();
      __syncthreads();

      FragBF bc, bn;
      load2(bc, brow);
#pragma unroll 4
      for (int k = 0; k < 512; k += 32) {
        if (k + 32 < 512) load2(bn, brow + k + 32);  // shadow-buffer B
        FragBF a;
        a.q[0] = *(const v4u*)(aldsrow + k);         // ds_load_b128
        a.q[1] = *(const v4u*)(aldsrow + k + 16);
        acc = WMMA_BF16(a.v, bc.v, acc);
        bc = bn;
      }
    }

    // Exchange gate tiles through LDS.
#pragma unroll
    for (int i = 0; i < 8; ++i) gb[gate][i + hi8][lr] = acc[i];
    __syncthreads();

    // Fused LSTM pointwise; c lives in registers across the whole scan.
    {
      const float ig = sigmoidf_(gb[0][pm0][pc0]);
      const float fg = sigmoidf_(gb[1][pm0][pc0]);
      const float gg = tanhf    (gb[2][pm0][pc0]);
      const float og = sigmoidf_(gb[3][pm0][pc0]);
      c0 = fg * c0 + ig * gg;
      const float h = og * tanhf(c0);
      const size_t hidx = (rowbase + m0 + pm0) * 512 + n0 + pc0;
      H[hidx] = f2bf(h);
      if (Fout) __builtin_nontemporal_store(h, Fout + hidx);
    }
    {
      const float ig = sigmoidf_(gb[0][pm1][pc1]);
      const float fg = sigmoidf_(gb[1][pm1][pc1]);
      const float gg = tanhf    (gb[2][pm1][pc1]);
      const float og = sigmoidf_(gb[3][pm1][pc1]);
      c1 = fg * c1 + ig * gg;
      const float h = og * tanhf(c1);
      const size_t hidx = (rowbase + m0 + pm1) * 512 + n0 + pc1;
      H[hidx] = f2bf(h);
      if (Fout) __builtin_nontemporal_store(h, Fout + hidx);
    }

    // Device-wide barrier: all h(t) stores visible before any block reads
    // them at t+1. Monotonic counter -> no reset races.
    __threadfence();
    __syncthreads();
    if (tid == 0) {
      __hip_atomic_fetch_add(bar, 1u, __ATOMIC_ACQ_REL, __HIP_MEMORY_SCOPE_AGENT);
      const unsigned target = nblocks * (unsigned)(t + 1);
      while (__hip_atomic_load(bar, __ATOMIC_ACQUIRE, __HIP_MEMORY_SCOPE_AGENT) < target)
        __builtin_amdgcn_s_sleep(2);
    }
    __syncthreads();
  }
}

// ---------------------------------------------------------------------------
// Host orchestration: 17 launches total, all on `stream` (graph-capture safe).
// ---------------------------------------------------------------------------
extern "C" void kernel_launch(void* const* d_in, const int* in_sizes, int n_in,
                              void* d_out, int out_size, void* d_ws, size_t ws_size,
                              hipStream_t stream) {
  (void)in_sizes; (void)n_in; (void)out_size; (void)ws_size;

  const float* x = (const float*)d_in[0];
  const float* Wih[3] = { (const float*)d_in[1], (const float*)d_in[5], (const float*)d_in[9]  };
  const float* Whh[3] = { (const float*)d_in[2], (const float*)d_in[6], (const float*)d_in[10] };
  const float* bih[3] = { (const float*)d_in[3], (const float*)d_in[7], (const float*)d_in[11] };
  const float* bhh[3] = { (const float*)d_in[4], (const float*)d_in[8], (const float*)d_in[12] };

  constexpr size_t SZW = 2048ull * 512;  // elements per weight matrix
  constexpr int    TB  = 16384;          // T*B flat rows

  // Workspace carve-up (all offsets 256B-aligned, ~172 MB total).
  char* ws = (char*)d_ws;
  unsigned short* WihB = (unsigned short*)(ws);               //  0        : 6 MB
  unsigned short* WhhB = (unsigned short*)(ws + 6291456);     //  6291456  : 6 MB
  float*          bias = (float*)         (ws + 12582912);    // 12582912  : 24 KB
  unsigned*       bar  = (unsigned*)      (ws + 12607488);    // 12607488  : 256 B
  unsigned short* Xa   = (unsigned short*)(ws + 12607744);    // 12607744  : 16 MB
  unsigned short* Xb   = (unsigned short*)(ws + 29384960);    // 29384960  : 16 MB
  float*          G    = (float*)         (ws + 46162176);    // 46162176  : 128 MB

  // 0) reset barrier counters (workspace is not re-zeroed between replays)
  k_zero_u32<<<1, 16, 0, stream>>>(bar, 16);

  // 1) precision conversion + bias fusion
  for (int l = 0; l < 3; ++l) {
    k_f32_to_bf16<<<(int)((SZW + 255) / 256), 256, 0, stream>>>(Wih[l], WihB + l * SZW, (int)SZW);
    k_f32_to_bf16<<<(int)((SZW + 255) / 256), 256, 0, stream>>>(Whh[l], WhhB + l * SZW, (int)SZW);
    k_bias_sum  <<<8, 256, 0, stream>>>(bih[l], bhh[l], bias + l * 2048, 2048);
  }
  k_f32_to_bf16<<<(TB * 512 + 255) / 256, 256, 0, stream>>>(x, Xa, TB * 512);

  // 2) per layer: hoisted input GEMM, then the persistent dilated scan
  unsigned short* Xin  = Xa;
  unsigned short* Xout = Xb;
  for (int l = 0; l < 3; ++l) {
    k_gemm_xw<<<dim3(TB / 32, 4), 256, 0, stream>>>(Xin, WihB + l * SZW, bias + l * 2048, G, TB);

    const int N  = 64 << l;    // rate * B rows per dilated step
    const int Td = 256 >> l;   // dilated steps
    float* fout = (l == 2) ? (float*)d_out : nullptr;
    k_lstm_layer<<<dim3(N / 16, 32), 128, 0, stream>>>(
        G, WhhB + l * SZW, Xout, fout, bar + l, N, Td);

    unsigned short* tmp = Xin; Xin = Xout; Xout = tmp;
  }
}
